// Decoder_87978110091940
// MI455X (gfx1250) — matrix operand order repaired and checked by reading, not compiled, not run on hardware
//
#include <hip/hip_runtime.h>
#include <hip/hip_bf16.h>
#include <math.h>

typedef __attribute__((ext_vector_type(16))) _Float16 v16h;
typedef __attribute__((ext_vector_type(8)))  float    v8f;

#define BB    8
#define TT    256
#define HD    256
#define ENCD  512
#define MELD  80
#define K1    864    // 592 (mel+ctx) + 256 (h1) = 848, padded to 27*32
#define KT1   27
#define K23   512    // 256 + 256
#define KT23  16
#define NTHR  512    // 16 wave32 waves

#if __has_builtin(__builtin_amdgcn_sched_barrier)
#define SCHED_FENCE() __builtin_amdgcn_sched_barrier(0)
#else
#define SCHED_FENCE() asm volatile("" ::: "memory")
#endif

// ---------------- prep kernels (loop-invariant hoisting) ----------------

// E[b*T+t][j] = sum_k enc[b,t,k] * Wa[j][256+k]   (encoder half of attention energy)
__global__ void prep_E(const float* __restrict__ enc, const float* __restrict__ Wa,
                       _Float16* __restrict__ E) {
  const int row = blockIdx.x;        // 0..2047
  const int j   = threadIdx.x;       // 0..255
  const float* er = enc + (size_t)row * ENCD;
  const float* wr = Wa + (size_t)j * (HD + ENCD) + HD;
  float acc = 0.f;
  #pragma unroll 4
  for (int k = 0; k < ENCD; ++k) acc += er[k] * wr[k];
  E[(size_t)row * HD + j] = (_Float16)acc;
}

__global__ void prep_ench(const float* __restrict__ enc, _Float16* __restrict__ out) {
  const size_t n = (size_t)BB * TT * ENCD;
  for (size_t i = blockIdx.x * blockDim.x + threadIdx.x; i < n; i += (size_t)gridDim.x * blockDim.x)
    out[i] = (_Float16)enc[i];
}

// Pack [Wih | Whh] (+zero pad) as row-major f16 so each LSTM is one WMMA GEMM.
__global__ void prep_pack(const float* Wih1, const float* Whh1,
                          const float* Wih2, const float* Whh2,
                          const float* Wih3, const float* Whh3,
                          _Float16* Wc1, _Float16* Wc2, _Float16* Wc3) {
  const int stride = gridDim.x * blockDim.x;
  const int tid = blockIdx.x * blockDim.x + threadIdx.x;
  for (int i = tid; i < 1024 * K1; i += stride) {
    int n = i / K1, k = i - n * K1;
    float v = (k < 592) ? Wih1[n * 592 + k]
            : (k < 848) ? Whh1[n * 256 + (k - 592)] : 0.f;
    Wc1[i] = (_Float16)v;
  }
  for (int i = tid; i < 1024 * K23; i += stride) {
    int n = i >> 9, k = i & 511;
    Wc2[i] = (_Float16)((k < 256) ? Wih2[n * 256 + k] : Whh2[n * 256 + (k - 256)]);
    Wc3[i] = (_Float16)((k < 256) ? Wih3[n * 256 + k] : Whh3[n * 256 + (k - 256)]);
  }
}

// ---------------- persistent fused decoder: 1 workgroup, 16 wave32 waves ----------------

__device__ __forceinline__ float sigm(float x) { return 1.0f / (1.0f + __expf(-x)); }

// Position of logical A-matrix element (m, k) inside the pre-swizzled fragment store:
// per K-tile (32 halves), lane = (bit4 of k%32)*16 + m holds 16 contiguous halves.
__device__ __forceinline__ int xpos(int m, int k) {
  int kt = k >> 5, r = k & 31;
  int lane = (((r >> 4) & 1) << 4) + m;
  int e = r & 15;
  return kt * 512 + lane * 16 + e;
}

__global__ __launch_bounds__(NTHR, 1)
void decoder_kernel(const int* __restrict__ pL,
                    const float* __restrict__ Wa, const float* __restrict__ ba,
                    const float* __restrict__ wv, const float* __restrict__ bv,
                    const float* __restrict__ bih1, const float* __restrict__ bhh1,
                    const float* __restrict__ bih2, const float* __restrict__ bhh2,
                    const float* __restrict__ bih3, const float* __restrict__ bhh3,
                    const float* __restrict__ Wm, const float* __restrict__ bm,
                    const float* __restrict__ Ws, const float* __restrict__ bs,
                    const _Float16* __restrict__ E, const _Float16* __restrict__ ench,
                    const _Float16* __restrict__ Wc1, const _Float16* __restrict__ Wc2,
                    const _Float16* __restrict__ Wc3,
                    float* __restrict__ out) {
  __shared__ __align__(64) _Float16 sXf[KT1 * 512];  // 27.6 KB A-fragments, WMMA layout
  __shared__ float    sG[8 * 1024];     // 32 KB    gate accumulators (WMMA f32 D)
  __shared__ float    sH[3][8 * HD];    // 24 KB    h1,h2,h3
  __shared__ float    sC[3][8 * HD];    // 24 KB    c1,c2,c3
  __shared__ _Float16 sCtx[8 * ENCD];   // 8 KB     attention context (f16)
  __shared__ float    sQP[8 * HD];      // 8 KB     query projection (+ba)
  __shared__ float    sSc[8 * TT];      // 8 KB     scores -> softmax weights
  __shared__ float    sPM[8 * MELD];    // 2.5 KB   previous mel frame

  const int tid  = threadIdx.x;
  const int wave = tid >> 5, lane = tid & 31;
  const int nl = lane & 15, hi = lane >> 4;
  const int L = *pL;

  // ---- init: zero state + fragment store (pad rows/cols stay zero forever)
  for (int i = tid; i < KT1 * 512; i += NTHR) sXf[i] = (_Float16)0.f;
  for (int i = tid; i < 8 * HD; i += NTHR) {
    sH[0][i] = sH[1][i] = sH[2][i] = 0.f;
    sC[0][i] = sC[1][i] = sC[2][i] = 0.f;
  }
  for (int i = tid; i < 8 * MELD; i += NTHR) sPM[i] = 0.f;
  __syncthreads();

  // WMMA GEMM: sG[0:8][0:1024] = X * W^T. K-tile outer: one LDS A-load feeds 4
  // independent WMMAs (N-tiles wave, wave+16, wave+32, wave+48). Each K-iteration
  // is fenced so the scheduler cannot hoist weight loads across iterations
  // (which previously exploded live ranges and spilled B-fragments to scratch).
  auto gemm = [&](const _Float16* __restrict__ W, int Kw, int ktiles) {
    v8f a0 = {}, a1 = {}, a2 = {}, a3 = {};
    const _Float16* w0 = W + (size_t)(wave * 16 + nl) * Kw + hi * 16;
    const _Float16* w1 = w0 + (size_t)256 * Kw;
    const _Float16* w2 = w0 + (size_t)512 * Kw;
    const _Float16* w3 = w0 + (size_t)768 * Kw;
    const _Float16* xp = sXf + lane * 16;
    #pragma clang loop unroll(disable)
    for (int kt = 0; kt < ktiles; ++kt) {
      v16h af = *(const v16h*)(xp + kt * 512);
      v16h b0 = *(const v16h*)(w0 + kt * 32);
      v16h b1 = *(const v16h*)(w1 + kt * 32);
      v16h b2 = *(const v16h*)(w2 + kt * 32);
      v16h b3 = *(const v16h*)(w3 + kt * 32);
      a0 = __builtin_amdgcn_wmma_f32_16x16x32_f16(false, af, false, b0, (short)0, a0, false, false);
      a1 = __builtin_amdgcn_wmma_f32_16x16x32_f16(false, af, false, b1, (short)0, a1, false, false);
      a2 = __builtin_amdgcn_wmma_f32_16x16x32_f16(false, af, false, b2, (short)0, a2, false, false);
      a3 = __builtin_amdgcn_wmma_f32_16x16x32_f16(false, af, false, b3, (short)0, a3, false, false);
      SCHED_FENCE();
    }
    if (hi == 0) {                    // D rows m=0..7 live in lanes 0..15, VGPR r -> row r
      const int c = wave * 16 + nl;
      #pragma unroll
      for (int r = 0; r < 8; ++r) {
        sG[r * 1024 + c]       = a0[r];
        sG[r * 1024 + c + 256] = a1[r];
        sG[r * 1024 + c + 512] = a2[r];
        sG[r * 1024 + c + 768] = a3[r];
      }
    }
  };

  auto lstm_ew = [&](float* h, float* c, const float* bi, const float* bh) {
    for (int idx = tid; idx < 8 * HD; idx += NTHR) {
      int b = idx >> 8, j = idx & 255;
      const float* g = sG + b * 1024;
      float gi = g[j]       + bi[j]       + bh[j];
      float gf = g[256 + j] + bi[256 + j] + bh[256 + j];
      float gg = g[512 + j] + bi[512 + j] + bh[512 + j];
      float go = g[768 + j] + bi[768 + j] + bh[768 + j];
      float cn = sigm(gf) * c[idx] + sigm(gi) * tanhf(gg);
      c[idx] = cn;
      h[idx] = sigm(go) * tanhf(cn);
    }
  };

  for (int s = 0; s < L; ++s) {
    // 1) query projection: qp[b][j] = h3[b]·Wa[j][0:256] + ba[j]
    for (int idx = tid; idx < 8 * HD; idx += NTHR) {
      int b = idx >> 8, j = idx & 255;
      const float* h3 = sH[2] + b * HD;
      const float* wr = Wa + (size_t)j * (HD + ENCD);
      float acc = ba[j];
      #pragma unroll 4
      for (int k = 0; k < HD; ++k) acc += h3[k] * wr[k];
      sQP[idx] = acc;
    }
    __syncthreads();

    // 2) scores[b][t] = wv · tanh(E[b,t,:] + qp[b,:]) + bv
    for (int idx = tid; idx < 8 * TT; idx += NTHR) {
      int b = idx >> 8, t = idx & 255;
      const _Float16* er = E + ((size_t)(b * TT + t)) * HD;
      const float* qp = sQP + b * HD;
      float acc = bv[0];
      #pragma unroll 4
      for (int j = 0; j < HD; ++j) acc += wv[j] * tanhf((float)er[j] + qp[j]);
      sSc[idx] = acc;
    }
    __syncthreads();

    // 3) softmax over T: one wave per batch row (wave32 shuffle reduction)
    if (wave < 8) {
      const int b = wave;
      float v[8], mx = -1e30f;
      #pragma unroll
      for (int q = 0; q < 8; ++q) { v[q] = sSc[b * TT + lane + q * 32]; mx = fmaxf(mx, v[q]); }
      for (int off = 16; off; off >>= 1) mx = fmaxf(mx, __shfl_xor(mx, off, 32));
      float sum = 0.f;
      #pragma unroll
      for (int q = 0; q < 8; ++q) { v[q] = __expf(v[q] - mx); sum += v[q]; }
      for (int off = 16; off; off >>= 1) sum += __shfl_xor(sum, off, 32);
      float inv = 1.0f / sum;
      #pragma unroll
      for (int q = 0; q < 8; ++q) sSc[b * TT + lane + q * 32] = v[q] * inv;
    }
    __syncthreads();

    // 4) context[b][d] = sum_t w[b][t] * enc[b][t][d]
    for (int idx = tid; idx < 8 * ENCD; idx += NTHR) {
      int b = idx >> 9, d = idx & 511;
      const _Float16* er = ench + ((size_t)b * TT) * ENCD + d;
      const float* w = sSc + b * TT;
      float acc = 0.f;
      #pragma unroll 4
      for (int t = 0; t < TT; ++t) acc += w[t] * (float)er[(size_t)t * ENCD];
      sCtx[idx] = (_Float16)acc;
    }
    __syncthreads();

    // 5) LSTM1: X = [prev_mel | context | h1]  (rows 0..7; pads stay zero)
    for (int idx = tid; idx < 8 * 848; idx += NTHR) {
      int b = idx / 848, k = idx - b * 848;
      float v = (k < MELD) ? sPM[b * MELD + k]
              : (k < 592)  ? (float)sCtx[b * ENCD + (k - MELD)]
                           : sH[0][b * HD + (k - 592)];
      sXf[xpos(b, k)] = (_Float16)v;
    }
    __syncthreads();
    gemm(Wc1, K1, KT1);  __syncthreads();
    lstm_ew(sH[0], sC[0], bih1, bhh1); __syncthreads();

    // 6) LSTM2:  X = [h1 | h2]
    for (int idx = tid; idx < 8 * K23; idx += NTHR) {
      int b = idx >> 9, k = idx & 511;
      float v = (k < 256) ? sH[0][b * HD + k] : sH[1][b * HD + (k - 256)];
      sXf[xpos(b, k)] = (_Float16)v;
    }
    __syncthreads();
    gemm(Wc2, K23, KT23); __syncthreads();
    lstm_ew(sH[1], sC[1], bih2, bhh2); __syncthreads();

    // 7) LSTM3:  X = [h2 | h3]
    for (int idx = tid; idx < 8 * K23; idx += NTHR) {
      int b = idx >> 9, k = idx & 511;
      float v = (k < 256) ? sH[1][b * HD + k] : sH[2][b * HD + (k - 256)];
      sXf[xpos(b, k)] = (_Float16)v;
    }
    __syncthreads();
    gemm(Wc3, K23, KT23); __syncthreads();
    lstm_ew(sH[2], sC[2], bih3, bhh3); __syncthreads();

    // 8) projections: dec=[h3|ctx]; mel -> out + prev_mel, stop -> out tail
    for (int idx = tid; idx < 8 * MELD + 8; idx += NTHR) {
      if (idx < 8 * MELD) {
        int b = idx / MELD, m = idx - b * MELD;
        const float* wr = Wm + (size_t)m * (HD + ENCD);
        const float* h3 = sH[2] + b * HD;
        const _Float16* cx = sCtx + b * ENCD;
        float acc = bm[m];
        #pragma unroll 4
        for (int k = 0; k < HD; ++k) acc += h3[k] * wr[k];
        #pragma unroll 4
        for (int d = 0; d < ENCD; ++d) acc += (float)cx[d] * wr[HD + d];
        out[((size_t)b * L + s) * MELD + m] = acc;
        sPM[b * MELD + m] = acc;
      } else {
        int b = idx - 8 * MELD;
        const float* h3 = sH[2] + b * HD;
        const _Float16* cx = sCtx + b * ENCD;
        float acc = bs[0];
        for (int k = 0; k < HD; ++k) acc += h3[k] * Ws[k];
        for (int d = 0; d < ENCD; ++d) acc += (float)cx[d] * Ws[HD + d];
        out[(size_t)BB * L * MELD + (size_t)b * L + s] = sigm(acc);
      }
    }
    __syncthreads();
  }
}

// ---------------- launch ----------------

extern "C" void kernel_launch(void* const* d_in, const int* in_sizes, int n_in,
                              void* d_out, int out_size, void* d_ws, size_t ws_size,
                              hipStream_t stream) {
  const float* enc  = (const float*)d_in[0];
  const int*   pL   = (const int*)d_in[1];
  const float* Wa   = (const float*)d_in[2];
  const float* ba   = (const float*)d_in[3];
  const float* wv   = (const float*)d_in[4];
  const float* bv   = (const float*)d_in[5];
  const float* Wih1 = (const float*)d_in[6];
  const float* Whh1 = (const float*)d_in[7];
  const float* bih1 = (const float*)d_in[8];
  const float* bhh1 = (const float*)d_in[9];
  const float* Wih2 = (const float*)d_in[10];
  const float* Whh2 = (const float*)d_in[11];
  const float* bih2 = (const float*)d_in[12];
  const float* bhh2 = (const float*)d_in[13];
  const float* Wih3 = (const float*)d_in[14];
  const float* Whh3 = (const float*)d_in[15];
  const float* bih3 = (const float*)d_in[16];
  const float* bhh3 = (const float*)d_in[17];
  const float* Wm   = (const float*)d_in[18];
  const float* bm   = (const float*)d_in[19];
  const float* Ws   = (const float*)d_in[20];
  const float* bs   = (const float*)d_in[21];
  float* out = (float*)d_out;

  // ws layout (f16 elements, all sub-buffers 32B-aligned)
  _Float16* wsh  = (_Float16*)d_ws;
  _Float16* E    = wsh;                        // 2048*256
  _Float16* ench = E    + (size_t)2048 * 256;  // 2048*512
  _Float16* Wc1  = ench + (size_t)2048 * 512;  // 1024*864
  _Float16* Wc2  = Wc1  + (size_t)1024 * K1;   // 1024*512
  _Float16* Wc3  = Wc2  + (size_t)1024 * K23;  // 1024*512

  prep_E<<<dim3(BB * TT), dim3(HD), 0, stream>>>(enc, Wa, E);
  prep_ench<<<dim3(1024), dim3(256), 0, stream>>>(enc, ench);
  prep_pack<<<dim3(512), dim3(256), 0, stream>>>(Wih1, Whh1, Wih2, Whh2, Wih3, Whh3,
                                                 Wc1, Wc2, Wc3);
  decoder_kernel<<<dim3(1), dim3(NTHR), 0, stream>>>(
      pL, Wa, ba, wv, bv, bih1, bhh1, bih2, bhh2, bih3, bhh3,
      Wm, bm, Ws, bs, E, ench, Wc1, Wc2, Wc3, out);
}
